// Aggregator_21122649161743
// MI455X (gfx1250) — matrix operand order repaired and checked by reading, not compiled, not run on hardware
//
#include <hip/hip_runtime.h>

// ---------------- types for WMMA / vector loads ----------------
typedef __attribute__((ext_vector_type(16))) __bf16 v16bf;
typedef __attribute__((ext_vector_type(8)))  float  v8f;
typedef __attribute__((ext_vector_type(8)))  unsigned int u32x8;
typedef __attribute__((ext_vector_type(4)))  float  f32x4;

#define DIM_D   512      // inner dim
#define DIM_DO  256      // output dim
#define NHEADS  8
#define NROWS   65536    // B*H
#define BLK_ROWS 64      // rows (B*H) per block
#define NKT     (DIM_D / 32)   // 16 K-tiles of 32
#define LDS_ROW_U32 260  // (512 bf16 * 2B + 16B pad) / 4  -> conflict-free b128 reads
#define BTILE_STRIDE (16 * (DIM_D / 2))   // +16 B-columns -> 4096 u32

// fp32 -> packed bf16x2, round-to-nearest-even
__device__ __forceinline__ unsigned int pack2bf(float x, float y) {
    unsigned int a = __builtin_bit_cast(unsigned int, x);
    unsigned int b = __builtin_bit_cast(unsigned int, y);
    a += 0x7FFFu + ((a >> 16) & 1u);
    b += 0x7FFFu + ((b >> 16) & 1u);
    return (a >> 16) | (b & 0xFFFF0000u);
}

__device__ __forceinline__ v16bf to16bf(uint4 lo, uint4 hi) {
    u32x8 u = {lo.x, lo.y, lo.z, lo.w, hi.x, hi.y, hi.z, hi.w};
    return __builtin_bit_cast(v16bf, u);
}

// ---------------- kernel 1: W fp32 -> bf16 (row-major, same layout) ----------------
__global__ __launch_bounds__(256) void wconv_kernel(const float* __restrict__ W,
                                                    unsigned int* __restrict__ Wbf) {
    int i = blockIdx.x * 256 + threadIdx.x;            // float4 index, 0..32767
    f32x4 v = ((const f32x4*)W)[i];
    uint2 o;
    o.x = pack2bf(v[0], v[1]);
    o.y = pack2bf(v[2], v[3]);
    ((uint2*)Wbf)[i] = o;
}

// ---------------- kernel 2: fused BN + swish + GEMM (WMMA bf16) + head-reduction ----------------
__global__ __launch_bounds__(256) void fused_kernel(
    const float* __restrict__ X,      // [65536, 512]
    const float* __restrict__ rel,    // [8192, 8]
    const float* __restrict__ gamma,
    const float* __restrict__ beta,
    const float* __restrict__ mean,
    const float* __restrict__ var,
    const unsigned int* __restrict__ Wbf,   // [256, 512] bf16 packed as u32 [256, 256]
    const float* __restrict__ bias,   // [256]
    float* __restrict__ out)          // [8192, 256]
{
    __shared__ unsigned int Alds[BLK_ROWS * LDS_ROW_U32];   // 66,560 B

    const int tid     = threadIdx.x;
    const int rowBase = blockIdx.x * BLK_ROWS;

    // ---- Phase 1: load X, BN + swish, bf16 -> LDS (row-major, padded) ----
    {
        const int c4 = tid & 127;    // float4 column index 0..127 (cols 4c..4c+3)
        const int rg = tid >> 7;     // 0..1 -> 32 rows each
        const f32x4 g4 = ((const f32x4*)gamma)[c4];
        const f32x4 b4 = ((const f32x4*)beta )[c4];
        const f32x4 m4 = ((const f32x4*)mean )[c4];
        const f32x4 v4 = ((const f32x4*)var  )[c4];
        f32x4 inv, sh;
        inv[0] = g4[0] * rsqrtf(v4[0] + 1e-5f);  sh[0] = b4[0] - m4[0] * inv[0];
        inv[1] = g4[1] * rsqrtf(v4[1] + 1e-5f);  sh[1] = b4[1] - m4[1] * inv[1];
        inv[2] = g4[2] * rsqrtf(v4[2] + 1e-5f);  sh[2] = b4[2] - m4[2] * inv[2];
        inv[3] = g4[3] * rsqrtf(v4[3] + 1e-5f);  sh[3] = b4[3] - m4[3] * inv[3];

        #pragma unroll 4
        for (int i = 0; i < 32; ++i) {
            const int r = rg * 32 + i;
            const f32x4* xp = (const f32x4*)(X + (size_t)(rowBase + r) * DIM_D) + c4;
            f32x4 x4 = __builtin_nontemporal_load(xp);   // X is streamed exactly once
            float y0 = x4[0] * inv[0] + sh[0];
            float y1 = x4[1] * inv[1] + sh[1];
            float y2 = x4[2] * inv[2] + sh[2];
            float y3 = x4[3] * inv[3] + sh[3];
            y0 = y0 / (1.0f + __expf(-y0));
            y1 = y1 / (1.0f + __expf(-y1));
            y2 = y2 / (1.0f + __expf(-y2));
            y3 = y3 / (1.0f + __expf(-y3));
            uint2 p;
            p.x = pack2bf(y0, y1);
            p.y = pack2bf(y2, y3);
            *(uint2*)&Alds[r * LDS_ROW_U32 + c4 * 2] = p;
        }
    }
    __syncthreads();

    // ---- Phase 2: WMMA GEMM; each wave = 4 M-tiles x 2 N-tiles (B frag reused 4x) ----
    const int lane  = tid & 31;
    const int wave  = tid >> 5;        // 0..7 -> N column group [wave*32, wave*32+32)
    const int lsub  = (lane >> 4) & 1; // half-wave select
    const int ncol  = lane & 15;

    v8f acc[4][2] = {};

    // A-fragment base: lane holds row (m*16 + ncol); lsub picks K phase (+16B)
    const unsigned int* aP = Alds + ncol * LDS_ROW_U32 + lsub * 4;
    // B-fragment base: lane holds column (wave*32 + j*16 + ncol), 16 K-values (2/u32)
    const unsigned int* bP = Wbf + (wave * 32 + ncol) * (DIM_D / 2) + lsub * 4;

    // One GEMM K-step. KT must be a literal so the WGP-scope prefetch offset is an
    // inline-asm immediate (no per-prefetch address VALU -> no WMMA/VALU hazard NOPs).
    // Scope omitted on global_prefetch_b8 => SCOPE_CU (WGP): fills WGP$/L0.
    #define GEMM_STEP(KT)                                                                 \
    do {                                                                                  \
        if ((KT) + 2 < NKT) {                                                             \
            asm volatile("global_prefetch_b8 %0, off offset:%1"                           \
                         :: "v"(bP), "n"((((KT) + 2) * 16) * 4));                         \
            asm volatile("global_prefetch_b8 %0, off offset:%1"                           \
                         :: "v"(bP), "n"(((((KT) + 2) * 16) + BTILE_STRIDE) * 4));        \
        }                                                                                 \
        uint4 bLo[2], bHi[2];                                                             \
        _Pragma("unroll")                                                                 \
        for (int j = 0; j < 2; ++j) {                                                     \
            bLo[j] = *(const uint4*)(bP + (KT) * 16 + j * BTILE_STRIDE);                  \
            bHi[j] = *(const uint4*)(bP + (KT) * 16 + j * BTILE_STRIDE + 8);              \
        }                                                                                 \
        uint4 aLo[4], aHi[4];                                                             \
        _Pragma("unroll")                                                                 \
        for (int m = 0; m < 4; ++m) {                                                     \
            const unsigned int* ap = aP + m * 16 * LDS_ROW_U32 + (KT) * 16;               \
            aLo[m] = *(const uint4*)(ap);                                                 \
            aHi[m] = *(const uint4*)(ap + 8);                                             \
        }                                                                                 \
        v16bf bf0 = to16bf(bLo[0], bHi[0]);                                               \
        v16bf bf1 = to16bf(bLo[1], bHi[1]);                                               \
        _Pragma("unroll")                                                                 \
        for (int m = 0; m < 4; ++m) {                                                     \
            const v16bf af = to16bf(aLo[m], aHi[m]);                                      \
            acc[m][0] = __builtin_amdgcn_wmma_f32_16x16x32_bf16(false, af, false, bf0,    \
                                                     (short)0, acc[m][0], false, false);  \
            acc[m][1] = __builtin_amdgcn_wmma_f32_16x16x32_bf16(false, af, false, bf1,    \
                                                     (short)0, acc[m][1], false, false);  \
        }                                                                                 \
        __builtin_amdgcn_sched_group_barrier(0x010, 4, 0);  /* VMEM loads */              \
        __builtin_amdgcn_sched_group_barrier(0x100, 8, 0);  /* DS reads   */              \
        __builtin_amdgcn_sched_group_barrier(0x008, 8, 0);  /* WMMA      */               \
    } while (0)

    GEMM_STEP(0);  GEMM_STEP(1);  GEMM_STEP(2);  GEMM_STEP(3);
    GEMM_STEP(4);  GEMM_STEP(5);  GEMM_STEP(6);  GEMM_STEP(7);
    GEMM_STEP(8);  GEMM_STEP(9);  GEMM_STEP(10); GEMM_STEP(11);
    GEMM_STEP(12); GEMM_STEP(13); GEMM_STEP(14); GEMM_STEP(15);
    #undef GEMM_STEP

    // ---- Phase 3: fold 8-head weighted reduction out of the C layout ----
    // acc[m][j] VGPR i: lanes 0-15 -> row m*16+i (batch bm), lanes 16-31 -> row m*16+8+i (batch bm+1)
    const int n0 = wave * 32 + ncol;           // N column of j=0 tile
    const float bias0 = bias[n0];
    const float bias1 = bias[n0 + 16];

    #pragma unroll
    for (int m = 0; m < 4; ++m) {
        const int bsel = blockIdx.x * 8 + m * 2 + lsub;
        const f32x4 w03 = ((const f32x4*)(rel + (size_t)bsel * NHEADS))[0];
        const f32x4 w47 = ((const f32x4*)(rel + (size_t)bsel * NHEADS))[1];
        const float srel = w03[0] + w03[1] + w03[2] + w03[3] + w47[0] + w47[1] + w47[2] + w47[3];
        float* orow = out + (size_t)bsel * DIM_DO;

        float s0 = bias0 * srel;
        float s1 = bias1 * srel;
        s0 += acc[m][0][0]*w03[0] + acc[m][0][1]*w03[1] + acc[m][0][2]*w03[2] + acc[m][0][3]*w03[3]
            + acc[m][0][4]*w47[0] + acc[m][0][5]*w47[1] + acc[m][0][6]*w47[2] + acc[m][0][7]*w47[3];
        s1 += acc[m][1][0]*w03[0] + acc[m][1][1]*w03[1] + acc[m][1][2]*w03[2] + acc[m][1][3]*w03[3]
            + acc[m][1][4]*w47[0] + acc[m][1][5]*w47[1] + acc[m][1][6]*w47[2] + acc[m][1][7]*w47[3];
        orow[n0]      = s0;
        orow[n0 + 16] = s1;
    }
}

extern "C" void kernel_launch(void* const* d_in, const int* in_sizes, int n_in,
                              void* d_out, int out_size, void* d_ws, size_t ws_size,
                              hipStream_t stream) {
    const float* X     = (const float*)d_in[0];
    const float* rel   = (const float*)d_in[1];
    const float* gamma = (const float*)d_in[2];
    const float* beta  = (const float*)d_in[3];
    const float* mean  = (const float*)d_in[4];
    const float* var   = (const float*)d_in[5];
    const float* W     = (const float*)d_in[6];
    const float* bias  = (const float*)d_in[7];
    float* out = (float*)d_out;

    // bf16 copy of W lives in workspace: 256*512*2 = 262,144 bytes
    unsigned int* Wbf = (unsigned int*)d_ws;

    wconv_kernel<<<128, 256, 0, stream>>>(W, Wbf);
    fused_kernel<<<NROWS / BLK_ROWS, 256, 0, stream>>>(X, rel, gamma, beta, mean, var,
                                                       Wbf, bias, out);
}
